// PRplusplusBlock_1589137900208
// MI455X (gfx1250) — matrix operand order repaired and checked by reading, not compiled, not run on hardware
//
#include <hip/hip_runtime.h>
#include <cstdint>
#include <cstddef>

// ---------------- problem constants ----------------
constexpr int H = 80, W = 96, T = 112;
constexpr int VOX = H * W * T;            // 860160
constexpr int TT  = T / 16;               // 7 voxel-tiles of 16 along T
constexpr int NTILE = H * W * TT;         // 53760 (divisible by 8)
constexpr int FH = H / 2, FW = W / 2, FT = T / 2;   // 40,48,56 (flow grid)
constexpr int PH = H + 4, PW = W + 4, PT = T + 4;   // padded box-sum dims
constexpr int PVOL = PH * PW * PT;

// padded activation space: zero halo of 1 (+2 extra H planes for the tap-overflow group)
constexpr int HP2 = H + 4;                // hp = h+kd, kd<=3
constexpr int WP2 = W + 2;
constexpr int TP2 = T + 2;
constexpr int PVOX = HP2 * WP2 * TP2;     // 938448

// ---------------- workspace layout (bytes) ----------------
// @0     : f16 weight matrices (<0.5MB)
// @1MiB  : padded f32 biases
// @2MiB  : flow_up f32 (10.4MB)
// @14MiB : xw f32 chan-major (55MB)          -> later hbuf f16 [PVOX][16] (30MB)
// @70MiB : pm f32 (55MB)                     -> later act1 f16 [PVOX][64] (120MB, spans pf) -> later res
// @126MiB: pf f32 (62.4MB)
// @190MiB: stack f16 [PVOX][64] (120MB)      -> later t2 f16 [PVOX][16]
constexpr size_t WGT_OFF  = 0;
constexpr size_t BIAS_OFF = (size_t)1   << 20;
constexpr size_t FLOW_OFF = (size_t)2   << 20;
constexpr size_t R1_OFF   = (size_t)14  << 20;
constexpr size_t R2_OFF   = (size_t)70  << 20;
constexpr size_t R3_OFF   = (size_t)126 << 20;
constexpr size_t R4_OFF   = (size_t)190 << 20;

typedef __attribute__((ext_vector_type(16))) _Float16 v16h;
typedef __attribute__((ext_vector_type(8)))  float    v8f;
typedef __attribute__((ext_vector_type(4)))  uint32_t v4u;

union AFrag { v16h h; v4u u[2]; };

__device__ __forceinline__ size_t pidx(int h, int w, int t) {
  // unpadded coords -> padded voxel index (+1 halo shift on each axis)
  return ((size_t)(h + 1) * WP2 + (w + 1)) * TP2 + (t + 1);
}

// ---------------- weight prep: OIDHW f32 -> [COPAD][KPAD] f16, K = tap*CINP + cin ----------------
__global__ void wprep_kernel(const float* __restrict__ w, _Float16* __restrict__ out,
                             int cout, int copad, int cinr, int cinp, int kpad) {
  int i = blockIdx.x * 256 + threadIdx.x;
  int total = copad * kpad;
  if (i >= total) return;
  int o = i / kpad;
  int K = i - o * kpad;
  int tap = K / cinp;            // cinp is 16 or 64 -> shift
  int cin = K - tap * cinp;
  float v = 0.0f;
  if (o < cout && tap < 27 && cin < cinr)
    v = w[((size_t)o * cinr + cin) * 27 + tap];
  out[i] = (_Float16)v;
}

// ---------------- bias prep: zero-pad to npad ----------------
__global__ void biasprep_kernel(const float* __restrict__ src, float* __restrict__ dst,
                                int n, int npad) {
  int i = threadIdx.x;
  if (i < npad) dst[i] = (i < n) ? src[i] : 0.0f;
}

// ---------------- zero the halo voxels of a padded channel-last activation tensor ----------------
template <int CINP>
__global__ void halo_zero_kernel(_Float16* __restrict__ buf) {
  int i = blockIdx.x * 256 + threadIdx.x;
  if (i >= PVOX) return;
  int tp = i % TP2;
  int wp = (i / TP2) % WP2;
  int hp = i / (TP2 * WP2);
  bool interior = (hp >= 1 && hp <= H) && (wp >= 1 && wp <= W) && (tp >= 1 && tp <= T);
  if (interior) return;
  _Float16* p = buf + (size_t)i * CINP;
  v16h z = {};
#pragma unroll
  for (int c = 0; c < CINP / 16; ++c) *(v16h*)(p + 16 * c) = z;
}

// ---------------- trilinear upsample (align_corners) of flow*2 ----------------
__global__ void upsample_flow_kernel(const float* __restrict__ flow, float* __restrict__ up) {
  int i = blockIdx.x * 256 + threadIdx.x;
  if (i >= 3 * VOX) return;
  int c   = i / VOX;
  int rem = i - c * VOX;
  int t = rem % T;
  int w = (rem / T) % W;
  int h = rem / (T * W);

  float ph = h * ((float)(FH - 1) / (float)(H - 1));
  float pw = w * ((float)(FW - 1) / (float)(W - 1));
  float pt = t * ((float)(FT - 1) / (float)(T - 1));
  int h0 = (int)floorf(ph); h0 = h0 < 0 ? 0 : (h0 > FH - 2 ? FH - 2 : h0);
  int w0 = (int)floorf(pw); w0 = w0 < 0 ? 0 : (w0 > FW - 2 ? FW - 2 : w0);
  int t0 = (int)floorf(pt); t0 = t0 < 0 ? 0 : (t0 > FT - 2 ? FT - 2 : t0);
  float fh = ph - (float)h0, fw = pw - (float)w0, ft = pt - (float)t0;

  const float* f = flow + (size_t)c * FH * FW * FT;
  auto at = [&](int a, int b, int d) -> float {
    return f[((size_t)a * FW + b) * FT + d];
  };
  float c00 = at(h0, w0,   t0) * (1.f - ft) + at(h0, w0,   t0+1) * ft;
  float c01 = at(h0, w0+1, t0) * (1.f - ft) + at(h0, w0+1, t0+1) * ft;
  float c10 = at(h0+1, w0,   t0) * (1.f - ft) + at(h0+1, w0,   t0+1) * ft;
  float c11 = at(h0+1, w0+1, t0) * (1.f - ft) + at(h0+1, w0+1, t0+1) * ft;
  float c0 = c00 * (1.f - fw) + c01 * fw;
  float c1 = c10 * (1.f - fw) + c11 * fw;
  up[i] = 2.0f * (c0 * (1.f - fh) + c1 * fh);
}

// ---------------- trilinear warp: xw f32 chan-major + stack interior ch0..15 ----------------
__global__ void warp_kernel(const float* __restrict__ x, const float* __restrict__ up,
                            float* __restrict__ xw, _Float16* __restrict__ stack) {
  int i = blockIdx.x * 256 + threadIdx.x;
  if (i >= 16 * VOX) return;
  int c = i / VOX;
  int v = i - c * VOX;
  int t = v % T;
  int w = (v / T) % W;
  int h = v / (T * W);

  float cx = (float)h + up[v];
  float cy = (float)w + up[VOX + v];
  float cz = (float)t + up[2 * VOX + v];
  float x0f = floorf(cx), y0f = floorf(cy), z0f = floorf(cz);
  int x0 = (int)x0f, y0 = (int)y0f, z0 = (int)z0f;
  float fx = cx - x0f, fy = cy - y0f, fz = cz - z0f;

  const float* src = x + (size_t)c * VOX;
  float acc = 0.0f;
#pragma unroll
  for (int dx = 0; dx < 2; ++dx)
#pragma unroll
    for (int dy = 0; dy < 2; ++dy)
#pragma unroll
      for (int dz = 0; dz < 2; ++dz) {
        int ix = x0 + dx, iy = y0 + dy, iz = z0 + dz;
        if ((unsigned)ix < (unsigned)H && (unsigned)iy < (unsigned)W && (unsigned)iz < (unsigned)T) {
          float wt = (dx ? fx : 1.f - fx) * (dy ? fy : 1.f - fy) * (dz ? fz : 1.f - fz);
          acc += wt * src[((size_t)ix * W + iy) * T + iz];
        }
      }
  xw[i] = acc;
  stack[pidx(h, w, t) * 64 + c] = (_Float16)acc;
}

// ---------------- y -> stack ch 43..58, zero-fill interior pad ch 59..63 ----------------
__global__ void pack_y_kernel(const float* __restrict__ y, _Float16* __restrict__ stack) {
  int i = blockIdx.x * 256 + threadIdx.x;
  if (i >= 21 * VOX) return;
  int c = i / VOX;                     // 0..20 -> stack channel 43+c
  int v = i - c * VOX;
  int t = v % T;
  int w = (v / T) % W;
  int h = v / (T * W);
  float val = (c < 16) ? y[(size_t)c * VOX + v] : 0.0f;
  stack[pidx(h, w, t) * 64 + 43 + c] = (_Float16)val;
}

// ---------------- 3^3 box sum of xw (same dims, zero pad) ----------------
__global__ void box_pm_kernel(const float* __restrict__ xw, float* __restrict__ pm) {
  int i = blockIdx.x * 256 + threadIdx.x;
  if (i >= 16 * VOX) return;
  int c = i / VOX;
  int v = i - c * VOX;
  int t = v % T;
  int w = (v / T) % W;
  int h = v / (T * W);
  const float* src = xw + (size_t)c * VOX;
  float s = 0.0f;
#pragma unroll
  for (int a = -1; a <= 1; ++a)
#pragma unroll
    for (int b = -1; b <= 1; ++b)
#pragma unroll
      for (int d = -1; d <= 1; ++d) {
        int hh = h + a, ww = w + b, tt = t + d;
        if ((unsigned)hh < (unsigned)H && (unsigned)ww < (unsigned)W && (unsigned)tt < (unsigned)T)
          s += src[((size_t)hh * W + ww) * T + tt];
      }
  pm[i] = s;
}

// ---------------- 3^3 box sum of y on padded grid ----------------
__global__ void box_pf_kernel(const float* __restrict__ y, float* __restrict__ pf) {
  int i = blockIdx.x * 256 + threadIdx.x;
  if (i >= 16 * PVOL) return;
  int c = i / PVOL;
  int v = i - c * PVOL;
  int d = v % PT;
  int b = (v / PT) % PW;
  int a = v / (PT * PW);
  int hc = a - 2, wc = b - 2, tc = d - 2;
  const float* src = y + (size_t)c * VOX;
  float s = 0.0f;
#pragma unroll
  for (int da = -1; da <= 1; ++da)
#pragma unroll
    for (int db = -1; db <= 1; ++db)
#pragma unroll
      for (int dd = -1; dd <= 1; ++dd) {
        int hh = hc + da, ww = wc + db, tt = tc + dd;
        if ((unsigned)hh < (unsigned)H && (unsigned)ww < (unsigned)W && (unsigned)tt < (unsigned)T)
          s += src[((size_t)hh * W + ww) * T + tt];
      }
  pf[i] = s;
}

// ---------------- correlation: 27 channels into stack ch 16..42 (f16, /27) ----------------
__global__ void corr_kernel(const float* __restrict__ pm, const float* __restrict__ pf,
                            _Float16* __restrict__ stack) {
  int vox = blockIdx.x * 256 + threadIdx.x;
  if (vox >= VOX) return;
  int t = vox % T;
  int w = (vox / T) % W;
  int h = vox / (T * W);
  float s[27];
#pragma unroll
  for (int q = 0; q < 27; ++q) s[q] = 0.0f;
  for (int c = 0; c < 16; ++c) {
    float m = pm[(size_t)c * VOX + vox];
    const float* pfc = pf + (size_t)c * PVOL;
#pragma unroll
    for (int i = 0; i < 3; ++i)
#pragma unroll
      for (int j = 0; j < 3; ++j)
#pragma unroll
        for (int k = 0; k < 3; ++k)
          s[i * 9 + j * 3 + k] += m * pfc[((size_t)(h + 2 * i) * PW + (w + 2 * j)) * PT + (t + 2 * k)];
  }
  const float inv = 1.0f / 27.0f;
  const size_t base = pidx(h, w, t) * 64 + 16;
#pragma unroll
  for (int q = 0; q < 27; ++q)
    stack[base + q] = (_Float16)(s[q] * inv);
}

// ---------------- implicit-GEMM conv via V_WMMA_F32_16X16X32_F16 ----------------
// act: padded channel-last [PVOX][CINP] f16, zero halo -> unconditional B loads.
// Wk : [COPAD][KPAD] f16, K = tap*CINP + cin; the block's 16-row panel is staged in LDS.
template <int CINP, int COP, bool RELU, bool HAS2, bool FINAL>
__global__ __launch_bounds__(256) void conv3d_wmma_kernel(
    const _Float16* __restrict__ act, const _Float16* __restrict__ act2,
    const _Float16* __restrict__ Wk, const float* __restrict__ biasp,
    _Float16* __restrict__ outh, float* __restrict__ outf) {
  constexpr int KPAD = (27 * CINP + 31) & ~31;         // 1728 (CINP=64) / 448 (CINP=16)
  constexpr int CSH  = (CINP == 64) ? 6 : 4;           // log2(CINP)

  const int co0 = blockIdx.y * 16;

  // ---- stage the 16 x KPAD weight panel in LDS (shared by all 8 waves) ----
  __shared__ _Float16 As[16 * KPAD];
  {
    const v4u* s = (const v4u*)(Wk + (size_t)co0 * KPAD);
    v4u* d = (v4u*)As;
#pragma unroll 2
    for (int j = threadIdx.x; j < 16 * KPAD / 8; j += 256) d[j] = s[j];
  }
  __syncthreads();

  const int lane = threadIdx.x & 31;
  const int wv   = threadIdx.x >> 5;
  const int tile = blockIdx.x * 8 + wv;
  if (tile >= NTILE) return;                           // wave-uniform (grid is exact)
  const int t0 = (tile % TT) * 16;
  const int w  = (tile / TT) % W;
  const int h  = tile / (TT * W);

  const int  n   = lane & 15;                          // B column / A row
  const bool hi  = lane >= 16;
  const int  kd0 = hi ? 4 : 0;                         // A b128 slot = kd0>>1: hi lanes take k 16..31
  const int  khi = hi ? 16 : 0;

  // padded-space base for kd=kh=kw=0 (i.e. hp=h, wp=w, tp=t0+n)
  const size_t bsp = (((size_t)h * WP2 + w) * TP2 + (t0 + n)) << CSH;

  v8f acc = {};
  for (int kc = 0; kc < KPAD; kc += 32) {
    // ---- A fragment from LDS: two ds_load_b128 ----
    AFrag a;
    const v4u* ap = (const v4u*)(As + (size_t)n * KPAD + kc);
    a.u[0] = ap[kd0 >> 1];
    a.u[1] = ap[1 + (kd0 >> 1)];

    // ---- B fragment: one tap, 16 consecutive cin, unconditional 32B load ----
    const int kb   = kc + khi;
    const int tap  = kb >> CSH;
    const int cinb = kb & (CINP - 1);
    const int kd = tap / 9;
    const int r2 = tap - kd * 9;
    const int kh = r2 / 3;
    const int kw = r2 - kh * 3;
    const size_t sp = bsp + (((size_t)((kd * WP2 + kh) * TP2 + kw)) << CSH) + cinb;

    v16h b = *(const v16h*)(act + sp);
    if (HAS2) b = b + *(const v16h*)(act2 + sp);       // v_pk_add_f16

    acc = __builtin_amdgcn_wmma_f32_16x16x32_f16(false, a.h, false, b, (short)0, acc,
                                                 false, false);
  }

  // ---- epilogue: D layout VGPR r -> M = r + (hi?8:0), N = n ----
  const int mbase = hi ? 8 : 0;
  const size_t pvox = pidx(h, w, t0 + n - 0);          // t = t0+n
  const int vox = (h * W + w) * T + t0 + n;
#pragma unroll
  for (int r = 0; r < 8; ++r) {
    int co = co0 + mbase + r;
    float v = acc[r] + biasp[co];
    if (RELU) v = fmaxf(v, 0.0f);
    if (FINAL) {
      if (co < 3) outf[(size_t)co * VOX + vox] = v;    // d_out is [3][VOX] f32
    } else {
      outh[pvox * COP + co] = (_Float16)v;             // padded channel-last
    }
  }
}

// ---------------- host launcher ----------------
extern "C" void kernel_launch(void* const* d_in, const int* in_sizes, int n_in,
                              void* d_out, int out_size, void* d_ws, size_t ws_size,
                              hipStream_t stream) {
  (void)in_sizes; (void)n_in; (void)out_size; (void)ws_size;
  const float* x    = (const float*)d_in[0];
  const float* y    = (const float*)d_in[1];
  const float* flow = (const float*)d_in[2];
  const float* w1a  = (const float*)d_in[3];
  const float* b1a  = (const float*)d_in[4];
  const float* w1b  = (const float*)d_in[5];
  const float* b1b  = (const float*)d_in[6];
  const float* w2a  = (const float*)d_in[7];
  const float* b2a  = (const float*)d_in[8];
  const float* w2b  = (const float*)d_in[9];
  const float* b2b  = (const float*)d_in[10];
  const float* wf   = (const float*)d_in[11];
  const float* bf   = (const float*)d_in[12];
  float* out = (float*)d_out;
  char*  ws  = (char*)d_ws;

  // f16 weight matrices (K = tap*CINP + cin)
  _Float16* Wg  = (_Float16*)(ws + WGT_OFF);
  _Float16* W1a = Wg;                  // 64 x 1728
  _Float16* W1b = Wg + 110592;         // 16 x 1728
  _Float16* W2a = Wg + 138240;         // 16 x 448
  _Float16* W2b = Wg + 145408;         // 16 x 448
  _Float16* Wfm = Wg + 152576;         // 16 x 448 (3 real rows)

  // padded biases
  float* Bp  = (float*)(ws + BIAS_OFF);
  float* B1a = Bp;        // 64
  float* B1b = Bp + 64;   // 16
  float* B2a = Bp + 80;   // 16
  float* B2b = Bp + 96;   // 16
  float* Bfm = Bp + 112;  // 16

  float*    flow_up = (float*)(ws + FLOW_OFF);
  float*    xw      = (float*)(ws + R1_OFF);
  float*    pm      = (float*)(ws + R2_OFF);
  float*    pf      = (float*)(ws + R3_OFF);
  _Float16* stack   = (_Float16*)(ws + R4_OFF);   // [PVOX][64]
  _Float16* act1    = (_Float16*)(ws + R2_OFF);   // [PVOX][64], reuse pm+pf after corr
  _Float16* hbuf    = (_Float16*)(ws + R1_OFF);   // [PVOX][16], reuse xw after box_pm
  _Float16* t2      = (_Float16*)(ws + R4_OFF);   // [PVOX][16], reuse stack after conv1a
  _Float16* res     = (_Float16*)(ws + R2_OFF);   // [PVOX][16], reuse act1 after conv1b

  // weight + bias prep
  wprep_kernel<<<(64 * 1728 + 255) / 256, 256, 0, stream>>>(w1a, W1a, 59, 64, 59, 64, 1728);
  wprep_kernel<<<(16 * 1728 + 255) / 256, 256, 0, stream>>>(w1b, W1b, 16, 16, 59, 64, 1728);
  wprep_kernel<<<(16 * 448 + 255) / 256, 256, 0, stream>>>(w2a, W2a, 16, 16, 16, 16, 448);
  wprep_kernel<<<(16 * 448 + 255) / 256, 256, 0, stream>>>(w2b, W2b, 16, 16, 16, 16, 448);
  wprep_kernel<<<(16 * 448 + 255) / 256, 256, 0, stream>>>(wf, Wfm, 3, 16, 16, 16, 448);
  biasprep_kernel<<<1, 64, 0, stream>>>(b1a, B1a, 59, 64);
  biasprep_kernel<<<1, 64, 0, stream>>>(b1b, B1b, 16, 16);
  biasprep_kernel<<<1, 64, 0, stream>>>(b2a, B2a, 16, 16);
  biasprep_kernel<<<1, 64, 0, stream>>>(b2b, B2b, 16, 16);
  biasprep_kernel<<<1, 64, 0, stream>>>(bf, Bfm, 3, 16);

  const int gpv = (PVOX + 255) / 256;

  // elementwise pipeline (stack halo first; producers fill interior)
  halo_zero_kernel<64><<<gpv, 256, 0, stream>>>(stack);
  upsample_flow_kernel<<<(3 * VOX + 255) / 256, 256, 0, stream>>>(flow, flow_up);
  warp_kernel<<<(16 * VOX + 255) / 256, 256, 0, stream>>>(x, flow_up, xw, stack);
  pack_y_kernel<<<(21 * VOX + 255) / 256, 256, 0, stream>>>(y, stack);
  box_pm_kernel<<<(16 * VOX + 255) / 256, 256, 0, stream>>>(xw, pm);
  box_pf_kernel<<<(16 * PVOL + 255) / 256, 256, 0, stream>>>(y, pf);
  corr_kernel<<<(VOX + 255) / 256, 256, 0, stream>>>(pm, pf, stack);

  // WMMA conv stack (halo-zero each destination after its region becomes free)
  dim3 blk(256);
  dim3 g1(NTILE / 8, 4);   // 64 output channels (59 real + zero pad)
  dim3 g2(NTILE / 8, 1);

  halo_zero_kernel<64><<<gpv, 256, 0, stream>>>(act1);   // after corr (region was pm/pf)
  halo_zero_kernel<16><<<gpv, 256, 0, stream>>>(hbuf);   // after box_pm/corr (region was xw)
  conv3d_wmma_kernel<64, 64, false, false, false><<<g1, blk, 0, stream>>>(stack, nullptr, W1a, B1a, act1, nullptr);
  halo_zero_kernel<16><<<gpv, 256, 0, stream>>>(t2);     // after conv1a (region was stack)
  conv3d_wmma_kernel<64, 16, true,  false, false><<<g2, blk, 0, stream>>>(act1, nullptr, W1b, B1b, hbuf, nullptr);
  halo_zero_kernel<16><<<gpv, 256, 0, stream>>>(res);    // after conv1b (region was act1)
  conv3d_wmma_kernel<16, 16, false, false, false><<<g2, blk, 0, stream>>>(hbuf, nullptr, W2a, B2a, t2, nullptr);
  conv3d_wmma_kernel<16, 16, true,  false, false><<<g2, blk, 0, stream>>>(t2, nullptr, W2b, B2b, res, nullptr);
  conv3d_wmma_kernel<16, 16, false, true,  true ><<<g2, blk, 0, stream>>>(hbuf, res, Wfm, Bfm, nullptr, out);
}
